// DGM_2293512536713
// MI455X (gfx1250) — matrix operand order repaired and checked by reading, not compiled, not run on hardware
//
#include <hip/hip_runtime.h>
#include <hip/hip_bf16.h>

typedef __attribute__((ext_vector_type(16))) _Float16 v16h;
typedef __attribute__((ext_vector_type(8)))  float    v8f;

#define DEVINL __device__ __forceinline__

constexpr int Bn = 4, Cc = 64, Hh = 128, Ww = 128, Kk = 3;
constexpr int HW = Hh * Ww;               // 16384
constexpr int NPIX = Bn * HW;             // 65536
constexpr int TILES = NPIX / 16;          // 4096 (16-pixel row tiles)
constexpr int WAVES = 4;
constexpr int TPB = WAVES * 32;           // 128 threads / block (4 wave32)
constexpr int NBLK = 256;                 // blocks; amortizes LDS weight preload
constexpr int TSTRIDE = NBLK * WAVES;     // tile stride (1024)
constexpr int TPW = TILES / TSTRIDE;      // 4 tiles per wave

// ---- CDNA5 WMMA 16x16x32 f16 (f32 accumulate) --------------------------
DEVINL v8f wmma32(v16h a, v16h b, v8f c) {
  return __builtin_amdgcn_wmma_f32_16x16x32_f16(
      /*neg_a=*/false, a, /*neg_b=*/false, b,
      /*c_mod=*/(short)0, c, /*reuse_a=*/false, /*reuse_b=*/false);
}

// A-matrix (16x32 f16) per-lane K mapping used here: contiguous, K = 16*(lane/16) + e
DEVINL int aK(int lane, int e) {
  return (lane < 16) ? e : e + 16;
}
// B-matrix (32x16 f16): lane<16 -> K=e, lane>=16 -> K=e+16

union Frag16 { v16h v; _Float16 h[16]; };

DEVINL float lrelu(float v) { return v >= 0.f ? v : 0.1f * v; }
DEVINL int iclampi(int v, int lo, int hi) { return v < lo ? lo : (v > hi ? hi : v); }

DEVINL void tile_coords(int tile, int lane, int& b, int& y, int& xw) {
  int gp = tile * 16 + (lane & 15);
  b  = gp >> 14;            // / (H*W)
  int rem = gp & (HW - 1);
  y  = rem >> 7;            // / W
  xw = rem & (Ww - 1);
}

// ========================================================================
// Kernel 1: offset conv  om = conv3x3(concat(x,inter), w_offset) -> dy,dx,mask
// GEMM: M=27(pad 32, 2 m-tiles), K=9 taps * 128 ch (36 chunks of 32), N=16 px
// ========================================================================
__global__ __launch_bounds__(TPB) void dgm_offset_kernel(
    const float* __restrict__ x, const float* __restrict__ inter,
    const float* __restrict__ w_offset, const float* __restrict__ b_offset,
    float* __restrict__ dyp, float* __restrict__ dxp, float* __restrict__ mkp)
{
  __shared__ __align__(32) _Float16 sw[72 * 512];   // 72KB: pre-swizzled A frags
  for (int i = threadIdx.x; i < 72 * 512; i += TPB) {
    int frag = i >> 9;
    int ln   = (i >> 4) & 31;
    int e    = i & 15;
    int mtile = frag & 1;
    int tc    = frag >> 1;            // tap*4 + chunk
    int tap = tc >> 2, chunk = tc & 3;
    int row = mtile * 16 + (ln & 15);
    int cin = chunk * 32 + aK(ln, e); // 0..127
    int kyy = tap / 3, kxx = tap % 3;
    float v = (row < 27) ? w_offset[((row * (2 * Cc) + cin) * Kk + kyy) * Kk + kxx] : 0.f;
    sw[i] = (_Float16)v;
  }
  __syncthreads();

  const int lane = threadIdx.x & 31;
  const int tile0 = blockIdx.x * WAVES + (threadIdx.x >> 5);

  for (int t = 0; t < TPW; ++t) {
    const int tile = tile0 + t * TSTRIDE;
    int b, y, xw; tile_coords(tile, lane, b, y, xw);

    v8f acc0 = {0,0,0,0,0,0,0,0};
    v8f acc1 = {0,0,0,0,0,0,0,0};

    for (int tap = 0; tap < 9; ++tap) {
      int yy = y + tap / 3 - 1;
      int xx = xw + tap % 3 - 1;
      bool inb = (yy >= 0) & (yy < Hh) & (xx >= 0) & (xx < Ww);
      #pragma unroll
      for (int chunk = 0; chunk < 4; ++chunk) {
        int c0 = chunk * 32 + (lane >> 4) * 16;          // concat channel base
        const float* src = (c0 < Cc) ? x : inter;
        int cc = c0 & (Cc - 1);
        Frag16 bf;
        if (inb) {
          long base = ((long)(b * Cc + cc) * Hh + yy) * Ww + xx;
          #pragma unroll
          for (int e = 0; e < 16; ++e) bf.h[e] = (_Float16)src[base + (long)e * HW];
        } else {
          #pragma unroll
          for (int e = 0; e < 16; ++e) bf.h[e] = (_Float16)0.f;
        }
        const _Float16* wbase = &sw[((tap * 4 + chunk) * 2) * 512 + lane * 16];
        v16h a0 = *(const v16h*)(wbase);
        v16h a1 = *(const v16h*)(wbase + 512);
        acc0 = wmma32(a0, bf.v, acc0);
        acc1 = wmma32(a1, bf.v, acc1);
      }
    }

    // D layout: row = mtile*16 + r + 8*(lane>=16), col = lane%16
    #pragma unroll
    for (int m = 0; m < 2; ++m) {
      v8f a = m ? acc1 : acc0;
      #pragma unroll
      for (int r = 0; r < 8; ++r) {
        int row = m * 16 + r + (lane >> 4) * 8;
        if (row < 27) {
          float v = a[r] + b_offset[row];
          if (row < 9) {
            dyp[((b * 9 + row) * Hh + y) * Ww + xw] = v;
          } else if (row < 18) {
            dxp[((b * 9 + (row - 9)) * Hh + y) * Ww + xw] = v;
          } else {
            mkp[((b * 9 + (row - 18)) * Hh + y) * Ww + xw] = 1.f / (1.f + __expf(-v));
          }
        }
      }
    }
  }
}

// ========================================================================
// Kernel 2: SFT  out = x + x*gamma + beta
// gamma = Wg2 * lrelu(Wg1 * inter), beta = Wb2 * lrelu(Wb1 * inter)
// ========================================================================
DEVINL void mm64(const _Float16* sw, int mat, Frag16 b0, Frag16 b1,
                 v8f acc[4], int lane) {
  #pragma unroll
  for (int m = 0; m < 4; ++m) {
    const _Float16* base = sw + mat * 4096 + (m * 2) * 512 + lane * 16;
    v16h a0 = *(const v16h*)(base);
    v16h a1 = *(const v16h*)(base + 512);
    acc[m] = wmma32(a0, b0.v, acc[m]);
    acc[m] = wmma32(a1, b1.v, acc[m]);
  }
}

// D-tile (4 m-tiles) -> B fragments for the next GEMM: one half-wave swap.
DEVINL void chainB(const v8f t[4], Frag16 bo[2], int lane) {
  bool lo = lane < 16;
  #pragma unroll
  for (int q = 0; q < 2; ++q) {
    #pragma unroll
    for (int r = 0; r < 8; ++r) {
      float vlo = t[2 * q][r];
      float vhi = t[2 * q + 1][r];
      float plo = __shfl_xor(vlo, 16, 32);
      float phi = __shfl_xor(vhi, 16, 32);
      bo[q].h[r]     = (_Float16)(lo ? vlo : phi);
      bo[q].h[8 + r] = (_Float16)(lo ? plo : vhi);
    }
  }
}

__global__ __launch_bounds__(TPB) void dgm_sft_kernel(
    const float* __restrict__ x, const float* __restrict__ inter,
    const float* __restrict__ wg1, const float* __restrict__ wg2,
    const float* __restrict__ wb1, const float* __restrict__ wb2,
    float* __restrict__ out)
{
  __shared__ __align__(32) _Float16 sw[4 * 8 * 512];  // 32KB: 4 matrices pre-swizzled
  {
    const float* mats[4] = {wg1, wg2, wb1, wb2};
    for (int i = threadIdx.x; i < 4 * 8 * 512; i += TPB) {
      int mat = i >> 12;
      int rest = i & 4095;
      int frag = rest >> 9;
      int ln   = (rest >> 4) & 31;
      int e    = rest & 15;
      int mtile = frag >> 1, chunk = frag & 1;
      int row = mtile * 16 + (ln & 15);
      int k   = chunk * 32 + aK(ln, e);
      sw[i] = (_Float16)mats[mat][row * Cc + k];
    }
  }
  __syncthreads();

  const int lane = threadIdx.x & 31;
  const int tile0 = blockIdx.x * WAVES + (threadIdx.x >> 5);

  for (int t = 0; t < TPW; ++t) {
    const int tile = tile0 + t * TSTRIDE;
    int b, y, xw; tile_coords(tile, lane, b, y, xw);

    // B fragments of the inter tile (64ch x 16px)
    Frag16 bi[2];
    #pragma unroll
    for (int chunk = 0; chunk < 2; ++chunk) {
      int c0 = chunk * 32 + (lane >> 4) * 16;
      long base = ((long)(b * Cc + c0) * Hh + y) * Ww + xw;
      #pragma unroll
      for (int e = 0; e < 16; ++e) bi[chunk].h[e] = (_Float16)inter[base + (long)e * HW];
    }

    v8f z = {0,0,0,0,0,0,0,0};
    v8f t1[4] = {z, z, z, z};
    mm64(sw, 0, bi[0], bi[1], t1, lane);           // Wg1 * inter
    #pragma unroll
    for (int m = 0; m < 4; ++m)
      #pragma unroll
      for (int r = 0; r < 8; ++r) t1[m][r] = lrelu(t1[m][r]);
    Frag16 bg[2]; chainB(t1, bg, lane);
    v8f gm[4] = {z, z, z, z};
    mm64(sw, 1, bg[0], bg[1], gm, lane);           // gamma

    v8f t2[4] = {z, z, z, z};
    mm64(sw, 2, bi[0], bi[1], t2, lane);           // Wb1 * inter
    #pragma unroll
    for (int m = 0; m < 4; ++m)
      #pragma unroll
      for (int r = 0; r < 8; ++r) t2[m][r] = lrelu(t2[m][r]);
    Frag16 bb[2]; chainB(t2, bb, lane);
    v8f bt[4] = {z, z, z, z};
    mm64(sw, 3, bb[0], bb[1], bt, lane);           // beta

    #pragma unroll
    for (int m = 0; m < 4; ++m) {
      #pragma unroll
      for (int r = 0; r < 8; ++r) {
        int o = m * 16 + r + (lane >> 4) * 8;
        long idx = ((long)(b * Cc + o) * Hh + y) * Ww + xw;
        float xv = x[idx];
        out[idx] = xv + xv * gm[m][r] + bt[m][r];
      }
    }
  }
}

// ========================================================================
// Kernel 3: DCN  out += einsum('bkhwc,ock->bohw', bilinear(x)*mask, w_dcn)
// GEMM: M=64 (4 m-tiles), K = 9 taps * 64 ch (18 chunks of 32), N=16 px
// ========================================================================
__global__ __launch_bounds__(TPB) void dgm_dcn_kernel(
    const float* __restrict__ x, const float* __restrict__ w_dcn,
    const float* __restrict__ dyp, const float* __restrict__ dxp,
    const float* __restrict__ mkp, float* __restrict__ out)
{
  __shared__ __align__(32) _Float16 sw[72 * 512];   // 72KB: pre-swizzled w_dcn frags
  for (int i = threadIdx.x; i < 72 * 512; i += TPB) {
    int frag = i >> 9;
    int ln   = (i >> 4) & 31;
    int e    = i & 15;
    int mtile = frag & 3;
    int tc    = frag >> 2;            // tap*2 + chunk
    int tap = tc >> 1, chunk = tc & 1;
    int o = mtile * 16 + (ln & 15);
    int c = chunk * 32 + aK(ln, e);
    sw[i] = (_Float16)w_dcn[(o * Cc + c) * 9 + tap];
  }
  __syncthreads();

  const int lane = threadIdx.x & 31;
  const int tile0 = blockIdx.x * WAVES + (threadIdx.x >> 5);

  for (int t = 0; t < TPW; ++t) {
    const int tile = tile0 + t * TSTRIDE;
    int b, y, xw; tile_coords(tile, lane, b, y, xw);
    const float* xb = x + (long)b * Cc * HW;

    v8f z = {0,0,0,0,0,0,0,0};
    v8f acc[4] = {z, z, z, z};

    for (int tap = 0; tap < 9; ++tap) {
      int pidx = ((b * 9 + tap) * Hh + y) * Ww + xw;
      float dy = dyp[pidx], dx = dxp[pidx], mk = mkp[pidx];
      float py = (float)y  + (float)(tap / 3 - 1) + dy;
      float px = (float)xw + (float)(tap % 3 - 1) + dx;
      float fy0 = floorf(py), fx0 = floorf(px);
      float wy = py - fy0, wx = px - fx0;
      int y0 = (int)fy0, x0 = (int)fx0;
      int y1 = y0 + 1,   x1 = x0 + 1;
      float vy0 = (y0 >= 0 && y0 < Hh) ? 1.f : 0.f;
      float vy1 = (y1 >= 0 && y1 < Hh) ? 1.f : 0.f;
      float vx0 = (x0 >= 0 && x0 < Ww) ? 1.f : 0.f;
      float vx1 = (x1 >= 0 && x1 < Ww) ? 1.f : 0.f;
      float w00 = (1.f - wy) * (1.f - wx) * mk * vy0 * vx0;
      float w01 = (1.f - wy) * wx         * mk * vy0 * vx1;
      float w10 = wy         * (1.f - wx) * mk * vy1 * vx0;
      float w11 = wy         * wx         * mk * vy1 * vx1;
      int yc0 = iclampi(y0, 0, Hh - 1), yc1 = iclampi(y1, 0, Hh - 1);
      int xc0 = iclampi(x0, 0, Ww - 1), xc1 = iclampi(x1, 0, Ww - 1);
      int o00 = yc0 * Ww + xc0, o01 = yc0 * Ww + xc1;
      int o10 = yc1 * Ww + xc0, o11 = yc1 * Ww + xc1;

      // prefetch next tap's offset/mask cachelines (global_prefetch_b8)
      if (tap < 8) {
        __builtin_prefetch(&dyp[pidx + HW], 0, 0);
        __builtin_prefetch(&dxp[pidx + HW], 0, 0);
        __builtin_prefetch(&mkp[pidx + HW], 0, 0);
      }

      #pragma unroll
      for (int chunk = 0; chunk < 2; ++chunk) {
        int c0 = chunk * 32 + (lane >> 4) * 16;
        Frag16 bf;
        #pragma unroll
        for (int e = 0; e < 16; ++e) {
          long cb = (long)(c0 + e) * HW;
          float s = w00 * xb[cb + o00] + w01 * xb[cb + o01]
                  + w10 * xb[cb + o10] + w11 * xb[cb + o11];
          bf.h[e] = (_Float16)s;
        }
        const _Float16* base = &sw[((tap * 2 + chunk) * 4) * 512 + lane * 16];
        #pragma unroll
        for (int m = 0; m < 4; ++m) {
          v16h a = *(const v16h*)(base + m * 512);
          acc[m] = wmma32(a, bf.v, acc[m]);
        }
      }
    }

    #pragma unroll
    for (int m = 0; m < 4; ++m) {
      #pragma unroll
      for (int r = 0; r < 8; ++r) {
        int o = m * 16 + r + (lane >> 4) * 8;
        long idx = ((long)(b * Cc + o) * Hh + y) * Ww + xw;
        out[idx] += acc[m][r];
      }
    }
  }
}

// ========================================================================
extern "C" void kernel_launch(void* const* d_in, const int* in_sizes, int n_in,
                              void* d_out, int out_size, void* d_ws, size_t ws_size,
                              hipStream_t stream) {
  const float* x        = (const float*)d_in[0];
  const float* inter    = (const float*)d_in[1];
  const float* w_offset = (const float*)d_in[2];
  const float* b_offset = (const float*)d_in[3];
  const float* w_dcn    = (const float*)d_in[4];
  const float* w_g1     = (const float*)d_in[5];
  const float* w_g2     = (const float*)d_in[6];
  const float* w_b1     = (const float*)d_in[7];
  const float* w_b2     = (const float*)d_in[8];
  float* out = (float*)d_out;

  // workspace: dy, dx, mask each [B,9,H,W] f32 -> 3 * 2.36 MB = 7.08 MB
  float* dyp = (float*)d_ws;
  float* dxp = dyp + (size_t)9 * NPIX;
  float* mkp = dxp + (size_t)9 * NPIX;

  dgm_offset_kernel<<<NBLK, TPB, 0, stream>>>(x, inter, w_offset, b_offset,
                                              dyp, dxp, mkp);
  dgm_sft_kernel<<<NBLK, TPB, 0, stream>>>(x, inter, w_g1, w_g2, w_b1, w_b2, out);
  dgm_dcn_kernel<<<NBLK, TPB, 0, stream>>>(x, w_dcn, dyp, dxp, mkp, out);
}
